// HGTModel_40037685133764
// MI455X (gfx1250) — matrix operand order repaired and checked by reading, not compiled, not run on hardware
//
#include <hip/hip_runtime.h>
#include <math.h>
#include <stdint.h>

// ---------------------------------------------------------------------------
// HGT forward for MI455X (gfx1250, wave32, WMMA).
// Dense GEMMs: v_wmma_f32_16x16x32_bf16, f32 accumulate, B panel staged
// global->LDS via GLOBAL_LOAD_ASYNC_TO_LDS_B32 (ASYNCcnt) and reused by all
// 8 waves of the block; each wave reuses its A fragment across 4 WMMAs.
// ---------------------------------------------------------------------------

#define NN   20000     // nodes per type (N_A == N_B)
#define CC   128       // channels
#define HH   8         // heads
#define EE   400000    // edges per edge type

typedef __attribute__((ext_vector_type(16))) __bf16 v16bf;
typedef __attribute__((ext_vector_type(8)))  float  v8f;

__device__ __forceinline__ float gelu_f(float x) {
  return 0.5f * x * (1.0f + erff(x * 0.70710678118654752f));
}

// ---------------------------------------------------------------------------
// C[M,N] = act(A[M,K]) @ B[K,N] + bias[N]   (row-major, bf16 WMMA, f32 acc)
// Block = 256 threads = 8 waves; block tile = 128(M) x 64(N); K-step = 32.
// ---------------------------------------------------------------------------
template <int ACT>
__global__ void gemm_bf16_wmma(const float* __restrict__ A,
                               const float* __restrict__ B,
                               const float* __restrict__ bias,
                               float* __restrict__ Cm,
                               int M, int N, int K)
{
  __shared__ float tileB[32 * 64];               // 8 KB fp32 B panel
  const int wave = threadIdx.x >> 5;
  const int lane = threadIdx.x & 31;
  const int r16  = lane & 15;
  const int hi   = lane >> 4;                    // lane group 0..15 / 16..31
  const int m0   = (blockIdx.x * 8 + wave) << 4;
  const int n0   = blockIdx.y << 6;
  const bool valid = (m0 < M);                   // wave-uniform

  v8f acc[4];
#pragma unroll
  for (int s = 0; s < 4; ++s) {
    const float bv = bias[n0 + s * 16 + r16];    // D elem r -> (M=r+8*hi, N): bias on N only
#pragma unroll
    for (int r = 0; r < 8; ++r) acc[s][r] = bv;
  }

  const float* Arow = valid ? (A + (size_t)(m0 + r16) * K + 8 * hi) : A;

  for (int k0 = 0; k0 < K; k0 += 32) {
    // ---- async global -> LDS fill of the 32x64 fp32 B panel (all waves) ----
    for (int idx = threadIdx.x; idx < 32 * 64; idx += 256) {
      const int kr = idx >> 6, cn = idx & 63;
      const float* gp = B + (size_t)(k0 + kr) * N + n0 + cn;
      const unsigned lo = (unsigned)(uintptr_t)(&tileB[idx]);   // LDS byte offset
      asm volatile("global_load_async_to_lds_b32 %0, %1, off"
                   :: "v"(lo), "v"(gp) : "memory");
    }
    asm volatile("s_wait_asynccnt 0x0" ::: "memory");
    __syncthreads();

    if (valid) {
      // A 16x32 bf16 fragment: element i -> K = i + (i&8) + 8*hi  (ISA 7.12.2)
      v16bf af;
#pragma unroll
      for (int i = 0; i < 16; ++i) {
        float av = Arow[k0 + i + (i & 8)];
        if (ACT == 1) av = gelu_f(av);
        af[i] = (__bf16)av;
      }
      // 4 x (B 32x16 fragment from LDS: element i -> K = i + (i&8) + 8*hi) -> 4 WMMA
#pragma unroll
      for (int s = 0; s < 4; ++s) {
        v16bf bf;
#pragma unroll
        for (int i = 0; i < 16; ++i)
          bf[i] = (__bf16)tileB[(i + (i & 8) + 8 * hi) * 64 + s * 16 + r16];
        acc[s] = __builtin_amdgcn_wmma_f32_16x16x32_bf16(false, af, false, bf,
                                                         (short)0, acc[s], false, false);
      }
    }
    __syncthreads();                             // panel consumed before refill
  }

  if (valid) {
#pragma unroll
    for (int s = 0; s < 4; ++s) {
      float* Crow = Cm + (size_t)(m0 + 8 * hi) * N + n0 + s * 16 + r16;
#pragma unroll
      for (int r = 0; r < 8; ++r) Crow[(size_t)r * N] = acc[s][r];
    }
  }
}

// ---------------------------------------------------------------------------
// Fold per-head rel transform into projection weight:
//   Wout[i, h*16+e] = sum_d W[i, h*16+d] * rel[h,d,e];  bout likewise for bias
// ---------------------------------------------------------------------------
__global__ void fuse_rel(const float* __restrict__ W, const float* __restrict__ bias,
                         const float* __restrict__ rel,
                         float* __restrict__ Wout, float* __restrict__ bout)
{
  int t = blockIdx.x * blockDim.x + threadIdx.x;
  if (t < CC * CC) {
    int i = t >> 7, c = t & 127, h = c >> 4, e = c & 15;
    const float* wr = W + (size_t)i * CC + h * 16;
    const float* rr = rel + h * 256 + e;
    float s = 0.f;
#pragma unroll
    for (int d = 0; d < 16; ++d) s += wr[d] * rr[d * 16];
    Wout[t] = s;
  } else if (t < CC * CC + CC) {
    int c = t - CC * CC, h = c >> 4, e = c & 15;
    float s = 0.f;
#pragma unroll
    for (int d = 0; d < 16; ++d) s += bias[h * 16 + d] * rel[h * 256 + d * 16 + e];
    bout[c] = s;
  }
}

// ---------------------------------------------------------------------------
// Edge attention: alpha + segment-max (monotone-uint float atomicMax)
// ---------------------------------------------------------------------------
__global__ void edge_alpha_max(const int* __restrict__ rows, const int* __restrict__ cols,
                               const float* __restrict__ q, const float* __restrict__ ke,
                               const float* __restrict__ p_rel,
                               float* __restrict__ alpha, unsigned* __restrict__ amaxU)
{
  int t = blockIdx.x * blockDim.x + threadIdx.x;
  if (t >= EE * HH) return;
  int e = t >> 3, h = t & 7;
  int r = rows[e], c = cols[e];
  const float* qp = q  + (size_t)c * CC + h * 16;
  const float* kp = ke + (size_t)r * CC + h * 16;
  float s = 0.f;
#pragma unroll
  for (int d = 0; d < 16; ++d) s += qp[d] * kp[d];
  s *= p_rel[h] * 0.25f;               // / sqrt(DH)
  alpha[t] = s;
  unsigned ub  = __float_as_uint(s);
  unsigned key = (ub & 0x80000000u) ? ~ub : (ub | 0x80000000u);
  atomicMax(&amaxU[(size_t)c * HH + h], key);
}

__global__ void edge_expden(const int* __restrict__ cols,
                            float* __restrict__ ex,            // in: alpha, out: exp
                            const unsigned* __restrict__ amaxU,
                            float* __restrict__ den)
{
  int t = blockIdx.x * blockDim.x + threadIdx.x;
  if (t >= EE * HH) return;
  int e = t >> 3, h = t & 7;
  int c = cols[e];
  unsigned key = amaxU[(size_t)c * HH + h];
  unsigned ub  = (key & 0x80000000u) ? (key & 0x7FFFFFFFu) : ~key;
  float am = __uint_as_float(ub);
  if (!(fabsf(am) < 3.0e38f)) am = 0.f;  // non-finite (empty segment) -> 0
  float v = expf(ex[t] - am);
  ex[t] = v;
  atomicAdd(&den[(size_t)c * HH + h], v);
}

__global__ void edge_scatter(const int* __restrict__ rows, const int* __restrict__ cols,
                             const float* __restrict__ ve, const float* __restrict__ ex,
                             const float* __restrict__ den, float* __restrict__ agg)
{
  int t = blockIdx.x * blockDim.x + threadIdx.x;
  if (t >= EE * CC) return;
  int e = t >> 7, ch = t & 127, h = ch >> 4;
  int r = rows[e], c = cols[e];
  float w = ex[(size_t)e * HH + h] / fmaxf(den[(size_t)c * HH + h], 1e-16f);
  atomicAdd(&agg[(size_t)c * CC + ch], ve[(size_t)r * CC + ch] * w);
}

// ---------------------------------------------------------------------------
// skip-gated mix, training-mode BN (batch stats), pooling, final dot
// ---------------------------------------------------------------------------
__global__ void skip_mix(const float* __restrict__ o, float* __restrict__ x,
                         const float* __restrict__ skip, int n)
{
  int t = blockIdx.x * blockDim.x + threadIdx.x;
  if (t >= n) return;
  float s = 1.0f / (1.0f + expf(-skip[0]));
  x[t] = s * o[t] + (1.0f - s) * x[t];
}

__global__ void bn_stats(const float* __restrict__ X, int M, int N,
                         float* __restrict__ mean, float* __restrict__ rinv)
{
  __shared__ float s1[256], s2[256];
  int col = blockIdx.x;
  float a = 0.f, b = 0.f;
  for (int r = threadIdx.x; r < M; r += 256) {
    float v = X[(size_t)r * N + col];
    a += v; b += v * v;
  }
  s1[threadIdx.x] = a; s2[threadIdx.x] = b;
  __syncthreads();
  for (int off = 128; off > 0; off >>= 1) {
    if (threadIdx.x < off) { s1[threadIdx.x] += s1[threadIdx.x + off];
                             s2[threadIdx.x] += s2[threadIdx.x + off]; }
    __syncthreads();
  }
  if (threadIdx.x == 0) {
    float mu  = s1[0] / (float)M;
    float var = s2[0] / (float)M - mu * mu;
    mean[col] = mu;
    rinv[col] = rsqrtf(fmaxf(var, 0.f) + 1e-5f);
  }
}

__global__ void bn_apply_relu(float* __restrict__ X, int total, int N,
                              const float* __restrict__ mean, const float* __restrict__ rinv,
                              const float* __restrict__ g, const float* __restrict__ b)
{
  int t = blockIdx.x * blockDim.x + threadIdx.x;
  if (t >= total) return;
  int col = t % N;
  float v = (X[t] - mean[col]) * rinv[col] * g[col] + b[col];
  X[t] = fmaxf(v, 0.f);
}

__global__ void avgpool2(const float* __restrict__ X, float* __restrict__ Y, int M)
{
  int t = blockIdx.x * blockDim.x + threadIdx.x;
  if (t >= M * 64) return;
  int r = t >> 6, j = t & 63;
  Y[t] = 0.5f * (X[(size_t)r * CC + 2 * j] + X[(size_t)r * CC + 2 * j + 1]);
}

__global__ void fc_out(const float* __restrict__ X, const float* __restrict__ Wv,
                       const float* __restrict__ b, float* __restrict__ out, int M)
{
  int r = blockIdx.x * blockDim.x + threadIdx.x;
  if (r >= M) return;
  float s = b[0];
#pragma unroll
  for (int j = 0; j < 64; ++j) s += X[(size_t)r * 64 + j] * Wv[j];
  out[r] = s;
}

// ---------------------------------------------------------------------------
// Orchestration.  Param leaf order = jax tree flatten (dict keys sorted):
//  0 x_a, 1 x_b, 2 edge0, 3 edge1,
//  4..7   bns: (g,b) x 2 layers
//  8..27  head: bn2..bn6 (g,b pairs), fc1_W/b .. fc5_W/b
//  28..51 layer0, 52..75 layer1 (a_rel e0/e1, k_W a/b, k_b a/b, m_rel e0/e1,
//         o_W a/b, o_b a/b, p_rel e0/e1, q_W a/b, q_b a/b, skip a/b, v_W a/b, v_b a/b)
// ---------------------------------------------------------------------------
extern "C" void kernel_launch(void* const* d_in, const int* in_sizes, int n_in,
                              void* d_out, int out_size, void* d_ws, size_t ws_size,
                              hipStream_t stream)
{
  (void)in_sizes; (void)n_in; (void)out_size; (void)ws_size;
  const size_t NC = (size_t)NN * CC;

  const float* in_xa = (const float*)d_in[0];
  const float* in_xb = (const float*)d_in[1];
  const int*   ed0   = (const int*)d_in[2];
  const int*   ed1   = (const int*)d_in[3];

  float* W  = (float*)d_ws;
  float* xa = W;                 // NC
  float* xb = W + NC;            // NC
  float* S  = W + 2 * NC;        // 8*NC shared region (attn bufs / head bufs)
  float *q_a = S,          *q_b = S + NC;
  float *ke0 = S + 2 * NC, *ve0 = S + 3 * NC;
  float *ke1 = S + 4 * NC, *ve1 = S + 5 * NC;
  float *agg_a = S + 6 * NC, *agg_b = S + 7 * NC;
  float* otmp = W + 10 * NC;                               // NC
  float* exb  = W + 11 * NC;                               // EE*HH
  unsigned* amaxU = (unsigned*)(exb + (size_t)EE * HH);    // NN*HH
  float* den   = (float*)(amaxU + (size_t)NN * HH);        // NN*HH
  float* Wf    = den + (size_t)NN * HH;                    // 4 * 128*128 fused weights
  float* bfv   = Wf + 4 * 16384;                           // 4 * 128 fused biases
  float* bnMean = bfv + 4 * 128;                           // 512
  float* bnRinv = bnMean + 512;                            // 512
  float* hb0 = S;                                          // head ping (<= NN*512)
  float* hb1 = S + 4 * NC;                                 // head pong

  hipMemcpyAsync(xa, in_xa, NC * sizeof(float), hipMemcpyDeviceToDevice, stream);
  hipMemcpyAsync(xb, in_xb, NC * sizeof(float), hipMemcpyDeviceToDevice, stream);

  auto gemm = [&](const float* A, const float* B, const float* bias, float* Cm,
                  int M, int N, int K, int act) {
    dim3 g((M + 127) / 128, N / 64), b(256);
    if (act) gemm_bf16_wmma<1><<<g, b, 0, stream>>>(A, B, bias, Cm, M, N, K);
    else     gemm_bf16_wmma<0><<<g, b, 0, stream>>>(A, B, bias, Cm, M, N, K);
  };
  auto bnpass = [&](float* X, int M, int N, const float* g, const float* b) {
    bn_stats<<<dim3(N), dim3(256), 0, stream>>>(X, M, N, bnMean, bnRinv);
    int total = M * N;
    bn_apply_relu<<<dim3((total + 255) / 256), dim3(256), 0, stream>>>(
        X, total, N, bnMean, bnRinv, g, b);
  };

  const int TEH = EE * HH;
  const size_t TEC = (size_t)EE * CC;

  for (int l = 0; l < 2; ++l) {
    int base = 28 + 24 * l;
    const float* a_rel0 = (const float*)d_in[base + 0];
    const float* a_rel1 = (const float*)d_in[base + 1];
    const float* kW_a   = (const float*)d_in[base + 2];
    const float* kW_b   = (const float*)d_in[base + 3];
    const float* kb_a   = (const float*)d_in[base + 4];
    const float* kb_b   = (const float*)d_in[base + 5];
    const float* m_rel0 = (const float*)d_in[base + 6];
    const float* m_rel1 = (const float*)d_in[base + 7];
    const float* oW_a   = (const float*)d_in[base + 8];
    const float* oW_b   = (const float*)d_in[base + 9];
    const float* ob_a   = (const float*)d_in[base + 10];
    const float* ob_b   = (const float*)d_in[base + 11];
    const float* p0     = (const float*)d_in[base + 12];
    const float* p1     = (const float*)d_in[base + 13];
    const float* qW_a   = (const float*)d_in[base + 14];
    const float* qW_b   = (const float*)d_in[base + 15];
    const float* qb_a   = (const float*)d_in[base + 16];
    const float* qb_b   = (const float*)d_in[base + 17];
    const float* sk_a   = (const float*)d_in[base + 18];
    const float* sk_b   = (const float*)d_in[base + 19];
    const float* vW_a   = (const float*)d_in[base + 20];
    const float* vW_b   = (const float*)d_in[base + 21];
    const float* vb_a   = (const float*)d_in[base + 22];
    const float* vb_b   = (const float*)d_in[base + 23];
    const float* bn_g   = (const float*)d_in[4 + 2 * l];
    const float* bn_b   = (const float*)d_in[5 + 2 * l];

    // fold a_rel / m_rel into k/v projection weights
    dim3 fg((CC * CC + CC + 255) / 256), fb(256);
    fuse_rel<<<fg, fb, 0, stream>>>(kW_a, kb_a, a_rel0, Wf + 0 * 16384, bfv + 0 * 128);
    fuse_rel<<<fg, fb, 0, stream>>>(vW_a, vb_a, m_rel0, Wf + 1 * 16384, bfv + 1 * 128);
    fuse_rel<<<fg, fb, 0, stream>>>(kW_b, kb_b, a_rel1, Wf + 2 * 16384, bfv + 2 * 128);
    fuse_rel<<<fg, fb, 0, stream>>>(vW_b, vb_b, m_rel1, Wf + 3 * 16384, bfv + 3 * 128);

    // projections (WMMA)
    gemm(xa, qW_a, qb_a, q_a, NN, CC, CC, 0);
    gemm(xb, qW_b, qb_b, q_b, NN, CC, CC, 0);
    gemm(xa, Wf + 0 * 16384, bfv + 0 * 128, ke0, NN, CC, CC, 0);
    gemm(xa, Wf + 1 * 16384, bfv + 1 * 128, ve0, NN, CC, CC, 0);
    gemm(xb, Wf + 2 * 16384, bfv + 2 * 128, ke1, NN, CC, CC, 0);
    gemm(xb, Wf + 3 * 16384, bfv + 3 * 128, ve1, NN, CC, CC, 0);

    hipMemsetAsync(agg_a, 0, 2 * NC * sizeof(float), stream);   // agg_a|agg_b contiguous

    // e0: a -> b  (dst = b)
    hipMemsetAsync(amaxU, 0, (size_t)NN * HH * 2 * sizeof(float), stream); // amax|den
    edge_alpha_max<<<dim3((TEH + 255) / 256), dim3(256), 0, stream>>>(
        ed0, ed0 + EE, q_b, ke0, p0, exb, amaxU);
    edge_expden<<<dim3((TEH + 255) / 256), dim3(256), 0, stream>>>(
        ed0 + EE, exb, amaxU, den);
    edge_scatter<<<dim3((unsigned)((TEC + 255) / 256)), dim3(256), 0, stream>>>(
        ed0, ed0 + EE, ve0, exb, den, agg_b);

    // e1: b -> a  (dst = a)
    hipMemsetAsync(amaxU, 0, (size_t)NN * HH * 2 * sizeof(float), stream);
    edge_alpha_max<<<dim3((TEH + 255) / 256), dim3(256), 0, stream>>>(
        ed1, ed1 + EE, q_a, ke1, p1, exb, amaxU);
    edge_expden<<<dim3((TEH + 255) / 256), dim3(256), 0, stream>>>(
        ed1 + EE, exb, amaxU, den);
    edge_scatter<<<dim3((unsigned)((TEC + 255) / 256)), dim3(256), 0, stream>>>(
        ed1, ed1 + EE, ve1, exb, den, agg_a);

    // out projection (GELU fused into A-load) + skip gate + BN + ReLU
    gemm(agg_a, oW_a, ob_a, otmp, NN, CC, CC, 1);
    skip_mix<<<dim3(((int)NC + 255) / 256), dim3(256), 0, stream>>>(otmp, xa, sk_a, (int)NC);
    bnpass(xa, NN, CC, bn_g, bn_b);

    gemm(agg_b, oW_b, ob_b, otmp, NN, CC, CC, 1);
    skip_mix<<<dim3(((int)NC + 255) / 256), dim3(256), 0, stream>>>(otmp, xb, sk_b, (int)NC);
    bnpass(xb, NN, CC, bn_g, bn_b);
  }

  // ---------------- MLP head (per node type, concatenated output) ----------
  const float *bn2g = (const float*)d_in[8],  *bn2b = (const float*)d_in[9];
  const float *bn3g = (const float*)d_in[10], *bn3b = (const float*)d_in[11];
  const float *bn4g = (const float*)d_in[12], *bn4b = (const float*)d_in[13];
  const float *bn5g = (const float*)d_in[14], *bn5b = (const float*)d_in[15];
  const float *bn6g = (const float*)d_in[16], *bn6b = (const float*)d_in[17];
  const float *fc1W = (const float*)d_in[18], *fc1b = (const float*)d_in[19];
  const float *fc2W = (const float*)d_in[20], *fc2b = (const float*)d_in[21];
  const float *fc3W = (const float*)d_in[22], *fc3b = (const float*)d_in[23];
  const float *fc4W = (const float*)d_in[24], *fc4b = (const float*)d_in[25];
  const float *fc5W = (const float*)d_in[26], *fc5b = (const float*)d_in[27];

  for (int t = 0; t < 2; ++t) {
    const float* xin = (t == 0) ? xa : xb;
    float* outp = (float*)d_out + (size_t)t * NN;
    const int M = NN;
    avgpool2<<<dim3((M * 64 + 255) / 256), dim3(256), 0, stream>>>(xin, hb0, M);
    bnpass(hb0, M, 64, bn2g, bn2b);
    gemm(hb0, fc1W, fc1b, hb1, M, 512, 64, 0);  bnpass(hb1, M, 512, bn3g, bn3b);
    gemm(hb1, fc2W, fc2b, hb0, M, 256, 512, 0); bnpass(hb0, M, 256, bn4g, bn4b);
    gemm(hb0, fc3W, fc3b, hb1, M, 128, 256, 0); bnpass(hb1, M, 128, bn5g, bn5b);
    gemm(hb1, fc4W, fc4b, hb0, M, 64, 128, 0);  bnpass(hb0, M, 64, bn6g, bn6b);
    fc_out<<<dim3((M + 255) / 256), dim3(256), 0, stream>>>(hb0, fc5W, fc5b, outp, M);
  }
}
